// WGCNDecoder_43241730736194
// MI455X (gfx1250) — matrix operand order repaired and checked by reading, not compiled, not run on hardware
//
#include <hip/hip_runtime.h>

// ---------------- problem constants ----------------
#define NODE_NUM      8040
#define NUM_GENE      8002
#define NUM_EDGE_PB   100000
#define NUM_GENE_EDGE 90000
#define BATCH         4
#define NTOT          (BATCH * NODE_NUM)     // 32160 rows
#define DIM           512
#define DEC           256
#define ETOT          (BATCH * NUM_EDGE_PB)  // 400000 edges
#define NF            ((long)NTOT * DIM)     // 16,465,920 feature elements
#define MTILES        (NTOT / 16)            // 2010 row tiles
#define BROWS         64                     // W rows staged per block
#define BSTRIDE       520                    // padded LDS row stride (bf16 elems)
#define SMEM_BYTES    (2 * BROWS * BSTRIDE * 2)  // hi+lo strips, 133,120 B

typedef __attribute__((ext_vector_type(16))) __bf16 v16bf;
typedef __attribute__((ext_vector_type(8)))  __bf16 v8bf;
typedef __attribute__((ext_vector_type(8)))  float  v8f;
typedef __attribute__((ext_vector_type(4)))  int    v4i;
typedef __attribute__((address_space(3)))    v4i    lds_v4i;

union bfpack {
    v16bf v;
    v8bf  h[2];
};

#if defined(__has_builtin)
#if __has_builtin(__builtin_amdgcn_global_load_async_to_lds_b128) && \
    __has_builtin(__builtin_amdgcn_s_wait_asynccnt)
#define USE_ASYNC_LDS 1
#endif
#endif

// ---------------- degree / coefficients ----------------
__global__ void k_deg_init(float* deg) {
    int i = blockIdx.x * blockDim.x + threadIdx.x;
    if (i < NTOT) deg[i] = 1.0f;                       // self loop contributes 1
}

__global__ void k_deg_count(const int* __restrict__ ei, float* deg) {
    int e = blockIdx.x * blockDim.x + threadIdx.x;
    if (e < ETOT) atomicAdd(&deg[ei[ETOT + e]], 1.0f); // dst row of edge_index
}

__global__ void k_dinv(float* d) {
    int i = blockIdx.x * blockDim.x + threadIdx.x;
    if (i < NTOT) d[i] = rsqrtf(d[i]);                 // in place: deg -> deg^-0.5
}

__global__ void k_selfcoef(const float* __restrict__ dinv,
                           const float* __restrict__ lw, float* selfc) {
    int n = blockIdx.x * blockDim.x + threadIdx.x;
    if (n < NTOT) {
        int   nm = n % NODE_NUM;
        float w  = (nm < NUM_GENE) ? lw[nm] : 1.0f;
        float di = dinv[n];
        selfc[n] = di * di * w;
    }
}

// ---------------- f32 -> split bf16 (hi + residual lo), optional ReLU ----------------
__global__ void k_split_bf16(const float* __restrict__ src,
                             __bf16* __restrict__ hi, __bf16* __restrict__ lo,
                             long n, int do_relu) {
    long i = (long)blockIdx.x * blockDim.x + threadIdx.x;
    if (i < n) {
        float v = src[i];
        if (do_relu) v = v > 0.0f ? v : 0.0f;
        __bf16 h = (__bf16)v;            // RNE
        float  r = v - (float)h;
        hi[i] = h;
        lo[i] = (__bf16)r;
    }
}

// ---------------- GEMM: XL = F @ W^T + b  (split-bf16, f32 accumulate) ----------------
// block = 8 waves; each wave computes a 16(M) x 64(N) strip; the 64-row W strip
// (hi+lo) is staged once per block into LDS (async-to-LDS path when available).
__global__ __launch_bounds__(256)
void k_gemm_wmma(const __bf16* __restrict__ Ahi, const __bf16* __restrict__ Alo,
                 const __bf16* __restrict__ Whi, const __bf16* __restrict__ Wlo,
                 const float*  __restrict__ bias, float* __restrict__ out) {
    extern __shared__ __bf16 smem[];
    __bf16* sBh = smem;                    // [BROWS][BSTRIDE]
    __bf16* sBl = smem + BROWS * BSTRIDE;

    const int tid  = threadIdx.x;
    const int wave = tid >> 5;
    const int lane = tid & 31;
    const int lq   = lane & 15;
    const int lh   = lane >> 4;
    const int n0   = blockIdx.y * 64;
    const int mt   = blockIdx.x * 8 + wave;      // 16-row tile index

    // ---- stage W rows n0..n0+63 (hi + lo) into LDS ----
    const __bf16* srcH = Whi + (long)n0 * DIM;
    const __bf16* srcL = Wlo + (long)n0 * DIM;
#if defined(USE_ASYNC_LDS)
    for (int c = tid; c < BROWS * (DIM / 8); c += 256) {
        const int r = c >> 6;
        const int o = (c & 63) * 8;
        __builtin_amdgcn_global_load_async_to_lds_b128(
            (v4i*)(srcH + (long)r * DIM + o),
            (lds_v4i*)(sBh + r * BSTRIDE + o), 0, 0);
        __builtin_amdgcn_global_load_async_to_lds_b128(
            (v4i*)(srcL + (long)r * DIM + o),
            (lds_v4i*)(sBl + r * BSTRIDE + o), 0, 0);
    }
    __builtin_amdgcn_s_wait_asynccnt(0);
#else
    for (int c = tid; c < BROWS * (DIM / 8); c += 256) {
        const int r = c >> 6;
        const int o = (c & 63) * 8;
        *(v8bf*)(sBh + r * BSTRIDE + o) = *(const v8bf*)(srcH + (long)r * DIM + o);
        *(v8bf*)(sBl + r * BSTRIDE + o) = *(const v8bf*)(srcL + (long)r * DIM + o);
    }
#endif
    __syncthreads();

    if (mt >= MTILES) return;                    // wave-uniform: EXEC stays all-1s
    const int m0 = mt * 16;

    v8f c[4];
    for (int t = 0; t < 4; ++t) c[t] = (v8f){};

    const __bf16* aph = Ahi + (long)(m0 + lq) * DIM;
    const __bf16* apl = Alo + (long)(m0 + lq) * DIM;

    for (int k0 = 0; k0 < DIM; k0 += 32) {
        // A 16x32 fragment: lane half selects K {0..7,16..23} vs {8..15,24..31}
        const int ka = k0 + 8 * lh;
        bfpack a_hi, a_lo;
        a_hi.h[0] = *(const v8bf*)(aph + ka);
        a_hi.h[1] = *(const v8bf*)(aph + ka + 16);
        a_lo.h[0] = *(const v8bf*)(apl + ka);
        a_lo.h[1] = *(const v8bf*)(apl + ka + 16);

        // B 32x16 fragment from LDS: lane = output column, K {0..7,16..23} vs {8..15,24..31}
        const int kb = k0 + 8 * lh;
        for (int t = 0; t < 4; ++t) {
            const __bf16* bh = sBh + (t * 16 + lq) * BSTRIDE + kb;
            const __bf16* bl = sBl + (t * 16 + lq) * BSTRIDE + kb;
            bfpack b_hi, b_lo;
            b_hi.h[0] = *(const v8bf*)(bh);
            b_hi.h[1] = *(const v8bf*)(bh + 16);
            b_lo.h[0] = *(const v8bf*)(bl);
            b_lo.h[1] = *(const v8bf*)(bl + 16);

            c[t] = __builtin_amdgcn_wmma_f32_16x16x32_bf16(
                       false, a_hi.v, false, b_hi.v, (short)0, c[t], false, false);
            c[t] = __builtin_amdgcn_wmma_f32_16x16x32_bf16(
                       false, a_hi.v, false, b_lo.v, (short)0, c[t], false, false);
            c[t] = __builtin_amdgcn_wmma_f32_16x16x32_bf16(
                       false, a_lo.v, false, b_hi.v, (short)0, c[t], false, false);
        }
    }

    // C layout: VGPR j -> row m0 + j + 8*lh, col n0 + 16t + lq
    for (int t = 0; t < 4; ++t) {
        const int   col = n0 + t * 16 + lq;
        const float bb  = bias[col];
        const int   rb  = m0 + 8 * lh;
        for (int j = 0; j < 8; ++j)
            out[(long)(rb + j) * DIM + col] = c[t][j] + bb;
    }
}

// ---------------- aggregation ----------------
// init AGG with the self-loop term (writes every element -> no memset needed)
__global__ void k_agg_init(const float* __restrict__ XL,
                           const float* __restrict__ selfc, float* __restrict__ AGG) {
    long i = (long)blockIdx.x * blockDim.x + threadIdx.x;
    if (i < NF) AGG[i] = selfc[i >> 9] * XL[i];
}

// one wave per edge: AGG[dst] += coef * XL[src]
__global__ __launch_bounds__(256)
void k_edge(const int* __restrict__ ei, const float* __restrict__ gw,
            const float* __restrict__ dinv, const float* __restrict__ XL,
            float* __restrict__ AGG) {
    int wave = (blockIdx.x * blockDim.x + threadIdx.x) >> 5;
    int lane = threadIdx.x & 31;
    if (wave >= ETOT) return;
    int   src  = ei[wave];
    int   dst  = ei[ETOT + wave];
    int   em   = wave % NUM_EDGE_PB;
    float w    = (em < NUM_GENE_EDGE) ? gw[em] : 1.0f;
    float coef = dinv[src] * dinv[dst] * w;
    const float* xs = XL + (long)src * DIM;
    float*       ad = AGG + (long)dst * DIM;
    for (int i = 0; i < 16; ++i) {
        int col = i * 32 + lane;                 // coalesced 128B per step
        atomicAdd(ad + col, coef * xs[col]);
    }
}

// ---------------- decoder: out[b] = (a P1) P2 (P1^T b) ----------------
__global__ __launch_bounds__(DEC)
void k_decoder(const float* __restrict__ AGG, const int* __restrict__ drug,
               const float* __restrict__ P1, const float* __restrict__ P2,
               float* __restrict__ out) {
    __shared__ float ua[DEC], ub[DEC], red[DEC];
    const int b = blockIdx.x;
    const int t = threadIdx.x;
    const int i0 = drug[2 * b] - 1;
    const int i1 = drug[2 * b + 1] - 1;
    const float* ra = AGG + (long)i0 * DIM;
    const float* rb = AGG + (long)i1 * DIM;
    float sa = 0.0f, sb = 0.0f;
    for (int e = 0; e < DIM; ++e) {
        float va = ra[e]; va = va > 0.0f ? va : 0.0f;   // ReLU of layer-3 output
        float vb = rb[e]; vb = vb > 0.0f ? vb : 0.0f;
        float p  = P1[e * DEC + t];
        sa += va * p;
        sb += vb * p;
    }
    ua[t] = sa; ub[t] = sb;
    __syncthreads();
    float s = 0.0f;
    for (int d = 0; d < DEC; ++d) s += ua[d] * P2[d * DEC + t];
    red[t] = s * ub[t];
    __syncthreads();
    for (int off = DEC / 2; off > 0; off >>= 1) {
        if (t < off) red[t] += red[t + off];
        __syncthreads();
    }
    if (t == 0) out[b] = red[0];
}

// ---------------- host orchestration ----------------
extern "C" void kernel_launch(void* const* d_in, const int* in_sizes, int n_in,
                              void* d_out, int out_size, void* d_ws, size_t ws_size,
                              hipStream_t stream) {
    const float* x    = (const float*)d_in[0];
    const int*   ei   = (const int*)d_in[1];
    const int*   drug = (const int*)d_in[2];
    const float* Wm[3] = {(const float*)d_in[4], (const float*)d_in[8],  (const float*)d_in[12]};
    const float* bv[3] = {(const float*)d_in[5], (const float*)d_in[9],  (const float*)d_in[13]};
    const float* gw[3] = {(const float*)d_in[6], (const float*)d_in[10], (const float*)d_in[14]};
    const float* lw[3] = {(const float*)d_in[7], (const float*)d_in[11], (const float*)d_in[15]};
    const float* P1 = (const float*)d_in[16];
    const float* P2 = (const float*)d_in[17];

    // workspace carve (256B aligned chunks)
    char* p = (char*)d_ws;
    auto carve = [&](size_t bytes) { char* q = p; p += (bytes + 255) & ~(size_t)255; return q; };
    float*  XL    = (float*)carve(NF * sizeof(float));
    float*  AGG   = (float*)carve(NF * sizeof(float));
    __bf16* Fhi   = (__bf16*)carve(NF * sizeof(__bf16));
    __bf16* Flo   = (__bf16*)carve(NF * sizeof(__bf16));
    __bf16* Whi   = (__bf16*)carve((size_t)DIM * DIM * sizeof(__bf16));
    __bf16* Wlo   = (__bf16*)carve((size_t)DIM * DIM * sizeof(__bf16));
    float*  dinv  = (float*)carve(NTOT * sizeof(float));
    float*  selfc = (float*)carve(NTOT * sizeof(float));

    const int  T       = 256;
    const int  gN      = (NTOT + T - 1) / T;
    const int  gE      = (ETOT + T - 1) / T;
    const long nWelem  = (long)DIM * DIM;
    const int  gWsplit = (int)((nWelem + T - 1) / T);
    const int  gFsplit = (int)((NF + T - 1) / T);

    // degree -> dinv (shared across layers)
    k_deg_init<<<gN, T, 0, stream>>>(dinv);
    k_deg_count<<<gE, T, 0, stream>>>(ei, dinv);
    k_dinv<<<gN, T, 0, stream>>>(dinv);

    const float* cur = x;
    int relu = 0;
    for (int l = 0; l < 3; ++l) {
        k_split_bf16<<<gFsplit, T, 0, stream>>>(cur, Fhi, Flo, NF, relu);
        k_split_bf16<<<gWsplit, T, 0, stream>>>(Wm[l], Whi, Wlo, nWelem, 0);
        dim3 gg((MTILES + 7) / 8, DIM / 64);
        k_gemm_wmma<<<gg, 256, SMEM_BYTES, stream>>>(Fhi, Flo, Whi, Wlo, bv[l], XL);
        k_selfcoef<<<gN, T, 0, stream>>>(dinv, lw[l], selfc);
        k_agg_init<<<gFsplit, T, 0, stream>>>(XL, selfc, AGG);
        k_edge<<<(ETOT * 32 + T - 1) / T, T, 0, stream>>>(ei, gw[l], dinv, XL, AGG);
        cur  = AGG;
        relu = 1;
    }

    k_decoder<<<BATCH, DEC, 0, stream>>>(AGG, drug, P1, P2, (float*)d_out);
}
